// AttentionReadout_32555852104145
// MI455X (gfx1250) — matrix operand order repaired and checked by reading, not compiled, not run on hardware
//
#include <hip/hip_runtime.h>
#include <hip/hip_bf16.h>
#include <cstdint>
#include <cstddef>

// ---------------- problem constants ----------------
constexpr int DD  = 768;     // model dim
constexpr int SS  = 1024;    // seq len
constexpr int BB  = 2;       // batch
constexpr int HH  = 8;       // heads
constexpr int HDD = 96;      // head dim
constexpr int MM  = 3072;    // ff dim
constexpr int VV  = 32000;   // vocab
constexpr int RR  = 2;       // residual blocks
constexpr int TTOK= 2 * SS + 1;  // 2049 concat tokens
constexpr int TP  = 2112;        // padded tokens (33 * 64)

typedef unsigned short u16;
typedef __attribute__((ext_vector_type(16))) __bf16 v16bf;
typedef __attribute__((ext_vector_type(8)))  float  v8f;

union FragAB { uint4 q[2]; v16bf v; };

__device__ __forceinline__ u16 f2bf(float f) {
    union { float f; uint32_t u; } c; c.f = f;
    uint32_t u = c.u;
    uint32_t r = u + 0x7FFFu + ((u >> 16) & 1u);
    return (u16)(r >> 16);
}

__device__ __forceinline__ float gelu_f(float x) {
    float c = 0.7978845608028654f * (x + 0.044715f * x * x * x);
    return 0.5f * x * (1.0f + tanhf(c));
}

// issue one 16B async copy global->LDS (GVS mode: saddr base + 32-bit vgpr offset)
__device__ __forceinline__ void async_copy_b128(uint32_t lds_addr, uint32_t gbyte_off,
                                                const void* base) {
    asm volatile("global_load_async_to_lds_b128 %0, %1, %2"
                 :: "v"(lds_addr), "v"(gbyte_off), "s"(base)
                 : "memory");
}

// ---------------- transpose + f32->bf16 weight prep ----------------
// in: K x N f32 (row-major), out: N x K bf16 (row-major) -- N-major weights
__global__ __launch_bounds__(256) void transpose_bf16_k(
    const float* __restrict__ in, u16* __restrict__ out, int K, int N)
{
    __shared__ float tile[32][33];
    const int n0 = blockIdx.x * 32, k0 = blockIdx.y * 32;
    const int tx = threadIdx.x & 31, ty = threadIdx.x >> 5;  // 32 x 8
#pragma unroll
    for (int r = 0; r < 4; r++)
        tile[ty * 4 + r][tx] = in[(size_t)(k0 + ty * 4 + r) * N + n0 + tx];
    __syncthreads();
#pragma unroll
    for (int r = 0; r < 4; r++)
        out[(size_t)(n0 + ty * 4 + r) * K + k0 + tx] = f2bf(tile[tx][ty * 4 + r]);
}

// ---------------- block reductions ----------------
__device__ __forceinline__ float wred_sum(float v) {
#pragma unroll
    for (int m = 16; m >= 1; m >>= 1) v += __shfl_xor(v, m, 32);
    return v;
}

// reduce 4 values across a 256-thread block; sh must hold >= 32 floats
__device__ __forceinline__ void block_reduce4(float& a, float& b, float& c, float& d, float* sh) {
    const int lane = threadIdx.x & 31, wid = threadIdx.x >> 5;
    a = wred_sum(a); b = wred_sum(b); c = wred_sum(c); d = wred_sum(d);
    if (lane == 0) { sh[wid] = a; sh[8 + wid] = b; sh[16 + wid] = c; sh[24 + wid] = d; }
    __syncthreads();
    if (threadIdx.x == 0) {
        float ta = 0, tb = 0, tc = 0, td = 0;
#pragma unroll
        for (int j = 0; j < 8; j++) { ta += sh[j]; tb += sh[8 + j]; tc += sh[16 + j]; td += sh[24 + j]; }
        sh[0] = ta; sh[8] = tb; sh[16] = tc; sh[24] = td;
    }
    __syncthreads();
    a = sh[0]; b = sh[8]; c = sh[16]; d = sh[24];
    __syncthreads();
}

// ---------------- fused dual-LayerNorm + residual input ----------------
__global__ __launch_bounds__(256) void ln_dual_k(
    const float* __restrict__ l2r, const float* __restrict__ r2l,
    const float* __restrict__ s1, const float* __restrict__ b1,
    const float* __restrict__ s2, const float* __restrict__ b2,
    float* __restrict__ x, u16* __restrict__ l2rn, u16* __restrict__ r2ln,
    u16* __restrict__ sumn)
{
    __shared__ float sh[32];
    const int row = blockIdx.x;
    const float* pa = l2r + (size_t)row * DD;
    const float* pb = r2l + (size_t)row * DD;
    float sa = 0, qa = 0, sb = 0, qb = 0;
    for (int d = threadIdx.x; d < DD; d += 256) {
        float a = pa[d], b = pb[d];
        sa += a; qa += a * a; sb += b; qb += b * b;
    }
    block_reduce4(sa, qa, sb, qb, sh);
    const float inv = 1.0f / DD;
    float m1 = sa * inv, m2 = sb * inv;
    float rs1 = rsqrtf(qa * inv - m1 * m1 + 1e-6f);
    float rs2 = rsqrtf(qb * inv - m2 * m2 + 1e-6f);
    for (int d = threadIdx.x; d < DD; d += 256) {
        float a = pa[d], b = pb[d];
        float an = (a - m1) * rs1 * s1[d] + b1[d];
        float bn = (b - m2) * rs2 * s2[d] + b2[d];
        size_t o = (size_t)row * DD + d;
        x[o] = a + b;
        l2rn[o] = f2bf(an);
        r2ln[o] = f2bf(bn);
        sumn[o] = f2bf(an + bn);
    }
}

// ---------------- generic LN -> bf16 ----------------
__global__ __launch_bounds__(256) void ln_bf16_k(
    const float* __restrict__ in, const float* __restrict__ s,
    const float* __restrict__ b, u16* __restrict__ outB)
{
    __shared__ float sh[32];
    const int row = blockIdx.x;
    const float* p = in + (size_t)row * DD;
    float sa = 0, qa = 0, d2 = 0, d3 = 0;
    for (int d = threadIdx.x; d < DD; d += 256) { float a = p[d]; sa += a; qa += a * a; }
    block_reduce4(sa, qa, d2, d3, sh);
    const float inv = 1.0f / DD;
    float m = sa * inv;
    float rs = rsqrtf(qa * inv - m * m + 1e-6f);
    for (int d = threadIdx.x; d < DD; d += 256)
        outB[(size_t)row * DD + d] = f2bf((p[d] - m) * rs * s[d] + b[d]);
}

// ---------------- LN + FiLM (x = a*LN(x)+b), writes f32 + bf16 ----------------
__global__ __launch_bounds__(256) void ln_film_k(
    float* __restrict__ x, const float* __restrict__ s, const float* __restrict__ b,
    const float* __restrict__ film, u16* __restrict__ outB)
{
    __shared__ float sh[32];
    const int row = blockIdx.x;
    const int bb = row / SS;
    float* p = x + (size_t)row * DD;
    float sa = 0, qa = 0, d2 = 0, d3 = 0;
    for (int d = threadIdx.x; d < DD; d += 256) { float a = p[d]; sa += a; qa += a * a; }
    block_reduce4(sa, qa, d2, d3, sh);
    const float inv = 1.0f / DD;
    float m = sa * inv;
    float rs = rsqrtf(qa * inv - m * m + 1e-6f);
    for (int d = threadIdx.x; d < DD; d += 256) {
        float v = (p[d] - m) * rs * s[d] + b[d];
        float fa = film[(size_t)bb * 2 * DD + d];
        float fb = film[(size_t)bb * 2 * DD + DD + d];
        float o = fa * v + fb;
        p[d] = o;
        outB[(size_t)row * DD + d] = f2bf(o);
    }
}

// ---------------- build padded concat [temb, l2r_n, r2l_n, 0...] bf16 ----------------
__global__ __launch_bounds__(256) void build_all_e_k(
    const float* __restrict__ temb, const u16* __restrict__ l2rn,
    const u16* __restrict__ r2ln, u16* __restrict__ aen)
{
    const long idx = (long)blockIdx.x * 256 + threadIdx.x;
    const long total = (long)BB * TP * DD;
    if (idx >= total) return;
    int b = (int)(idx / ((long)TP * DD));
    long r = idx % ((long)TP * DD);
    int t = (int)(r / DD), d = (int)(r % DD);
    u16 val;
    if (t == 0)            val = f2bf(temb[(size_t)b * DD + d]);
    else if (t <= SS)      val = l2rn[((size_t)b * SS + (t - 1)) * DD + d];
    else if (t <= 2 * SS)  val = r2ln[((size_t)b * SS + (t - SS - 1)) * DD + d];
    else                   val = 0;
    aen[idx] = val;
}

// ---------------- WMMA GEMM: C[Mr x N] = A[Mr x K](bf16) * Wt[N x K]^T + bias ----------------
// Async double-buffered LDS staging via global_load_async_to_lds_b128 (ASYNCcnt).
// EPI: 0 f32 store | 1 bf16*scale | 2 relu->bf16 | 3 gelu->bf16
//      4 f32 resid add (+opt bf16 mirror) | 5 bf16 transposed [b][n][t] store
template<int EPI>
__global__ __launch_bounds__(256) void gemm_k(
    const u16* __restrict__ A, const u16* __restrict__ Wt,
    const float* __restrict__ bias, int N, int K,
    float* __restrict__ outF, u16* __restrict__ outB,
    const float* __restrict__ resid, float scale, int TpLd)
{
    // ping-pong buffers: [buf][A(128x72) | B(128x72)]
    __shared__ u16 smem[2 * 2 * 128 * 72];
    constexpr int TILE = 128 * 72;          // u16 elements per matrix tile
    constexpr int BUF  = 2 * TILE;          // u16 elements per buffer
    const int tid = threadIdx.x;
    const int wave = tid >> 5, lane = tid & 31;
    const int ln15 = lane & 15, hi = lane >> 4;
    const int wr = wave >> 1, wc = wave & 1;
    const int m0 = blockIdx.y * 128, n0 = blockIdx.x * 128;

    // this thread's 4 staging chunks (row, 16B column slot)
    int srow[4], sco[4];
#pragma unroll
    for (int i = 0; i < 4; i++) {
        int c = tid + i * 256;
        srow[i] = c >> 3;
        sco[i]  = (c & 7) * 8;
    }
    const uint32_t ldsBase = (uint32_t)(uintptr_t)&smem[0];

    auto stage_async = [&](int kk, int buf) {
#pragma unroll
        for (int i = 0; i < 4; i++) {
            uint32_t aoff = (uint32_t)(((size_t)(m0 + srow[i]) * K + kk + sco[i]) * 2);
            uint32_t boff = (uint32_t)(((size_t)(n0 + srow[i]) * K + kk + sco[i]) * 2);
            uint32_t la = ldsBase + (uint32_t)((buf * BUF + srow[i] * 72 + sco[i]) * 2);
            uint32_t lb = ldsBase + (uint32_t)((buf * BUF + TILE + srow[i] * 72 + sco[i]) * 2);
            async_copy_b128(la, aoff, A);
            async_copy_b128(lb, boff, Wt);
        }
    };

    v8f acc[2][4];
#pragma unroll
    for (int a = 0; a < 2; a++)
#pragma unroll
        for (int b = 0; b < 4; b++)
#pragma unroll
            for (int i = 0; i < 8; i++) acc[a][b][i] = 0.0f;

    stage_async(0, 0);
    int buf = 0;
    for (int kk = 0; kk < K; kk += 64) {
        if (kk + 64 < K) {
            stage_async(kk + 64, buf ^ 1);
            // previous 8 copies must have landed; the 8 just issued may remain in flight
            asm volatile("s_wait_asynccnt 0x8" ::: "memory");
        } else {
            asm volatile("s_wait_asynccnt 0x0" ::: "memory");
        }
        __syncthreads();
        const u16* sA = &smem[buf * BUF];
        const u16* sB = &smem[buf * BUF + TILE];
#pragma unroll
        for (int kc = 0; kc < 2; kc++) {
            FragAB af[2], bf[4];
#pragma unroll
            for (int mi = 0; mi < 2; mi++) {
                int base = (wr * 32 + mi * 16 + ln15) * 72 + kc * 32 + hi * 8;
                af[mi].q[0] = *reinterpret_cast<const uint4*>(&sA[base]);
                af[mi].q[1] = *reinterpret_cast<const uint4*>(&sA[base + 16]);
            }
#pragma unroll
            for (int nj = 0; nj < 4; nj++) {
                int base = (wc * 64 + nj * 16 + ln15) * 72 + kc * 32 + hi * 8;
                bf[nj].q[0] = *reinterpret_cast<const uint4*>(&sB[base]);
                bf[nj].q[1] = *reinterpret_cast<const uint4*>(&sB[base + 16]);
            }
#pragma unroll
            for (int mi = 0; mi < 2; mi++)
#pragma unroll
                for (int nj = 0; nj < 4; nj++)
                    acc[mi][nj] = __builtin_amdgcn_wmma_f32_16x16x32_bf16(
                        false, af[mi].v, false, bf[nj].v, (short)0, acc[mi][nj], false, false);
        }
        __syncthreads();
        buf ^= 1;
    }

#pragma unroll
    for (int mi = 0; mi < 2; mi++)
#pragma unroll
        for (int nj = 0; nj < 4; nj++) {
            int col = n0 + wc * 64 + nj * 16 + ln15;
            float bv = bias[col];
#pragma unroll
            for (int i = 0; i < 8; i++) {
                int row = m0 + wr * 32 + mi * 16 + hi * 8 + i;
                float v = acc[mi][nj][i] + bv;
                if (EPI == 0) {
                    outF[(size_t)row * N + col] = v;
                } else if (EPI == 1) {
                    outB[(size_t)row * N + col] = f2bf(v * scale);
                } else if (EPI == 2) {
                    outB[(size_t)row * N + col] = f2bf(v > 0.0f ? v : 0.0f);
                } else if (EPI == 3) {
                    outB[(size_t)row * N + col] = f2bf(gelu_f(v));
                } else if (EPI == 4) {
                    float o = v + resid[(size_t)row * N + col];
                    outF[(size_t)row * N + col] = o;
                    if (outB) outB[(size_t)row * N + col] = f2bf(o);
                } else if (EPI == 5) {
                    int b = row / TpLd, t = row % TpLd;
                    outB[((size_t)b * N + col) * TpLd + t] = f2bf(v);
                }
            }
        }
}

// ---------------- WMMA flash attention (async double-buffered K/V staging) ----------------
// q:   [B*S, D] bf16 (pre-scaled by 1/sqrt(HD)), col = h*HD+hd
// k:   [B*TP, D] bf16
// vT:  [B, D, TP] bf16  (transposed V)
// out: [B*S, D] bf16
__global__ __launch_bounds__(128) void flash_k(
    const u16* __restrict__ q, const u16* __restrict__ kbuf,
    const u16* __restrict__ vT, u16* __restrict__ xatt)
{
    constexpr int KT_U16 = 64 * 104;            // K tile [key][d], stride 104
    constexpr int VT_U16 = 96 * 72;             // V tile [d][key], stride 72
    constexpr int FBUF   = KT_U16 + VT_U16;     // one ping-pong buffer
    __shared__ u16 fsmem[2 * FBUF];
    __shared__ u16 sP[4 * 16 * 64];             // per-wave probs [row][key]
    const int tid = threadIdx.x;
    const int wave = tid >> 5, lane = tid & 31;
    const int ln15 = lane & 15, hi = lane >> 4;
    const int qt = blockIdx.x, h = blockIdx.y, b = blockIdx.z;
    const int pbase = wave * 16 * 64;
    const uint32_t ldsBase = (uint32_t)(uintptr_t)&fsmem[0];

    // this thread's 6 staging chunks for each matrix
    auto stage_async = [&](int kt, int bufsel) {
#pragma unroll
        for (int i = 0; i < 6; i++) {          // K tile: 64 rows x 12 chunks
            int c = tid + i * 128;
            int row = c / 12, co = (c % 12) * 8;
            uint32_t la = ldsBase + (uint32_t)((bufsel * FBUF + row * 104 + co) * 2);
            uint32_t go = (uint32_t)(((size_t)(b * TP + kt * 64 + row) * DD + h * HDD + co) * 2);
            async_copy_b128(la, go, kbuf);
        }
#pragma unroll
        for (int i = 0; i < 6; i++) {          // V tile: 96 rows x 8 chunks
            int c = tid + i * 128;
            int row = c >> 3, co = (c & 7) * 8;
            uint32_t la = ldsBase + (uint32_t)((bufsel * FBUF + KT_U16 + row * 72 + co) * 2);
            uint32_t go = (uint32_t)((((size_t)b * DD + h * HDD + row) * TP + kt * 64 + co) * 2);
            async_copy_b128(la, go, vT);
        }
    };

    FragAB aq[3];
    {
        const size_t qrow = (size_t)(b * SS + qt * 64 + wave * 16 + ln15) * DD + h * HDD;
#pragma unroll
        for (int kc = 0; kc < 3; kc++) {
            aq[kc].q[0] = *reinterpret_cast<const uint4*>(q + qrow + kc * 32 + hi * 8);
            aq[kc].q[1] = *reinterpret_cast<const uint4*>(q + qrow + kc * 32 + 16 + hi * 8);
        }
    }
    float mrow[8], lrow[8];
    v8f o[6];
#pragma unroll
    for (int i = 0; i < 8; i++) { mrow[i] = -3.0e38f; lrow[i] = 0.0f; }
#pragma unroll
    for (int nf = 0; nf < 6; nf++)
#pragma unroll
        for (int i = 0; i < 8; i++) o[nf][i] = 0.0f;

    stage_async(0, 0);
    int cur = 0;
    for (int kt = 0; kt < TP / 64; kt++) {
        if (kt + 1 < TP / 64) {
            stage_async(kt + 1, cur ^ 1);
            // previous tile's 12 copies must have landed; the new 12 may remain in flight
            asm volatile("s_wait_asynccnt 0xc" ::: "memory");
        } else {
            asm volatile("s_wait_asynccnt 0x0" ::: "memory");
        }
        __syncthreads();
        const u16* sK = &fsmem[cur * FBUF];
        const u16* sV = &fsmem[cur * FBUF + KT_U16];

        // S = Q K^T (16 x 64)
        v8f sc[4];
#pragma unroll
        for (int nj = 0; nj < 4; nj++)
#pragma unroll
            for (int i = 0; i < 8; i++) sc[nj][i] = 0.0f;
#pragma unroll
        for (int kc = 0; kc < 3; kc++)
#pragma unroll
            for (int nj = 0; nj < 4; nj++) {
                FragAB bk;
                int base = (nj * 16 + ln15) * 104 + kc * 32 + hi * 8;
                bk.q[0] = *reinterpret_cast<const uint4*>(&sK[base]);
                bk.q[1] = *reinterpret_cast<const uint4*>(&sK[base + 16]);
                sc[nj] = __builtin_amdgcn_wmma_f32_16x16x32_bf16(
                    false, aq[kc].v, false, bk.v, (short)0, sc[nj], false, false);
            }

        // mask + online softmax
        const int qb0 = qt * 64 + wave * 16 + hi * 8;
        float tmax[8];
#pragma unroll
        for (int i = 0; i < 8; i++) tmax[i] = -3.0e38f;
#pragma unroll
        for (int nj = 0; nj < 4; nj++) {
            int j = kt * 64 + nj * 16 + ln15;
#pragma unroll
            for (int i = 0; i < 8; i++) {
                int qi = qb0 + i;
                bool vis = (j == 0) || (j <= SS && qi >= j - 1) ||
                           (j > SS && j <= 2 * SS && qi <= j - SS - 1);
                float s = vis ? sc[nj][i] : -3.0e38f;
                sc[nj][i] = s;
                tmax[i] = fmaxf(tmax[i], s);
            }
        }
#pragma unroll
        for (int i = 0; i < 8; i++) {
#pragma unroll
            for (int msk = 8; msk >= 1; msk >>= 1)
                tmax[i] = fmaxf(tmax[i], __shfl_xor(tmax[i], msk, 32));
        }
        float scl[8];
#pragma unroll
        for (int i = 0; i < 8; i++) {
            float mn = fmaxf(mrow[i], tmax[i]);
            scl[i] = __expf(mrow[i] - mn);
            mrow[i] = mn;
        }
        float rs[8];
#pragma unroll
        for (int i = 0; i < 8; i++) rs[i] = 0.0f;
#pragma unroll
        for (int nj = 0; nj < 4; nj++)
#pragma unroll
            for (int i = 0; i < 8; i++) {
                float p = __expf(sc[nj][i] - mrow[i]);
                rs[i] += p;
                sP[pbase + (i + 8 * hi) * 64 + nj * 16 + ln15] = f2bf(p);
            }
#pragma unroll
        for (int i = 0; i < 8; i++) {
#pragma unroll
            for (int msk = 8; msk >= 1; msk >>= 1)
                rs[i] += __shfl_xor(rs[i], msk, 32);
            lrow[i] = lrow[i] * scl[i] + rs[i];
        }
#pragma unroll
        for (int nf = 0; nf < 6; nf++)
#pragma unroll
            for (int i = 0; i < 8; i++) o[nf][i] *= scl[i];

        __syncthreads();  // make sP visible as uint4 reads

        // O += P V (16 x 96)
#pragma unroll
        for (int kc2 = 0; kc2 < 2; kc2++) {
            FragAB ap;
            int abase = pbase + ln15 * 64 + kc2 * 32 + hi * 8;
            ap.q[0] = *reinterpret_cast<const uint4*>(&sP[abase]);
            ap.q[1] = *reinterpret_cast<const uint4*>(&sP[abase + 16]);
#pragma unroll
            for (int nf = 0; nf < 6; nf++) {
                FragAB bv;
                int bbase = (nf * 16 + ln15) * 72 + kc2 * 32 + hi * 8;
                bv.q[0] = *reinterpret_cast<const uint4*>(&sV[bbase]);
                bv.q[1] = *reinterpret_cast<const uint4*>(&sV[bbase + 16]);
                o[nf] = __builtin_amdgcn_wmma_f32_16x16x32_bf16(
                    false, ap.v, false, bv.v, (short)0, o[nf], false, false);
            }
        }
        __syncthreads();  // all P.V reads done before next tile's async copies overwrite
        cur ^= 1;
    }

#pragma unroll
    for (int nf = 0; nf < 6; nf++) {
        int col = h * HDD + nf * 16 + ln15;
#pragma unroll
        for (int i = 0; i < 8; i++) {
            int row = b * SS + qt * 64 + wave * 16 + hi * 8 + i;
            xatt[(size_t)row * DD + col] = f2bf(o[nf][i] / lrow[i]);
        }
    }
}

// ---------------- tiny temb path (B=2): scalar f32 kernels ----------------
__global__ __launch_bounds__(256) void temb_mlp1_k(
    const float* __restrict__ temb, const float* __restrict__ w1,
    const float* __restrict__ b1, float* __restrict__ t1)
{
    int idx = blockIdx.x * 256 + threadIdx.x;
    if (idx >= BB * MM) return;
    int b = idx / MM, j = idx % MM;
    float a = b1[j];
    for (int d = 0; d < DD; d++) a += temb[(size_t)b * DD + d] * w1[(size_t)d * MM + j];
    t1[idx] = gelu_f(a);
}

__global__ __launch_bounds__(256) void temb_mlp2_k(
    const float* __restrict__ t1, const float* __restrict__ w2,
    const float* __restrict__ b2, float* __restrict__ t)
{
    int idx = blockIdx.x * 256 + threadIdx.x;
    if (idx >= BB * 4 * DD) return;
    int b = idx / (4 * DD), j = idx % (4 * DD);
    float a = b2[j];
    for (int k = 0; k < MM; k++) a += t1[(size_t)b * MM + k] * w2[(size_t)k * 4 * DD + j];
    t[idx] = a;
}

__global__ __launch_bounds__(256) void film_kern(
    const float* __restrict__ t, const float* __restrict__ fw,
    const float* __restrict__ fb, float* __restrict__ film)
{
    int idx = blockIdx.x * 256 + threadIdx.x;
    if (idx >= RR * BB * 2 * DD) return;
    int i = idx / (BB * 2 * DD);
    int rem = idx % (BB * 2 * DD);
    int b = rem / (2 * DD), j = rem % (2 * DD);
    float a = fb[(size_t)i * 2 * DD + j];
    for (int k = 0; k < 4 * DD; k++)
        a += t[(size_t)b * 4 * DD + k] * fw[(size_t)i * 4 * DD * 2 * DD + (size_t)k * 2 * DD + j];
    film[idx] = a;
}

// ---------------- host-side launch ----------------
static void launch_gemm(int epi, int Mr, int N, int K, hipStream_t s,
                        const u16* A, const u16* Wt, const float* bias,
                        float* outF, u16* outB, const float* resid,
                        float scale, int tp)
{
    dim3 g(N / 128, Mr / 128), blk(256);
    switch (epi) {
        case 0: gemm_k<0><<<g, blk, 0, s>>>(A, Wt, bias, N, K, outF, outB, resid, scale, tp); break;
        case 1: gemm_k<1><<<g, blk, 0, s>>>(A, Wt, bias, N, K, outF, outB, resid, scale, tp); break;
        case 2: gemm_k<2><<<g, blk, 0, s>>>(A, Wt, bias, N, K, outF, outB, resid, scale, tp); break;
        case 3: gemm_k<3><<<g, blk, 0, s>>>(A, Wt, bias, N, K, outF, outB, resid, scale, tp); break;
        case 4: gemm_k<4><<<g, blk, 0, s>>>(A, Wt, bias, N, K, outF, outB, resid, scale, tp); break;
        case 5: gemm_k<5><<<g, blk, 0, s>>>(A, Wt, bias, N, K, outF, outB, resid, scale, tp); break;
    }
}

extern "C" void kernel_launch(void* const* d_in, const int* in_sizes, int n_in,
                              void* d_out, int out_size, void* d_ws, size_t ws_size,
                              hipStream_t stream)
{
    const float* l2r    = (const float*)d_in[0];
    const float* r2l    = (const float*)d_in[1];
    const float* temb   = (const float*)d_in[2];
    const float* lnl2rs = (const float*)d_in[3];
    const float* lnl2rb = (const float*)d_in[4];
    const float* lnr2ls = (const float*)d_in[5];
    const float* lnr2lb = (const float*)d_in[6];
    const float* Wq     = (const float*)d_in[7];
    const float* bq     = (const float*)d_in[8];
    const float* Wk     = (const float*)d_in[9];
    const float* bk     = (const float*)d_in[10];
    const float* Wv     = (const float*)d_in[11];
    const float* bv     = (const float*)d_in[12];
    const float* Wo     = (const float*)d_in[13];
    const float* bo     = (const float*)d_in[14];
    const float* ffls   = (const float*)d_in[15];
    const float* fflb   = (const float*)d_in[16];
    const float* ffw1   = (const float*)d_in[17];
    const float* ffb1   = (const float*)d_in[18];
    const float* ffw2   = (const float*)d_in[19];
    const float* ffb2   = (const float*)d_in[20];
    const float* tmw1   = (const float*)d_in[21];
    const float* tmb1   = (const float*)d_in[22];
    const float* tmw2   = (const float*)d_in[23];
    const float* tmb2   = (const float*)d_in[24];
    const float* filmw  = (const float*)d_in[25];
    const float* filmb  = (const float*)d_in[26];
    const float* rw1    = (const float*)d_in[27];
    const float* rb1    = (const float*)d_in[28];
    const float* rw2    = (const float*)d_in[29];
    const float* rb2    = (const float*)d_in[30];
    const float* rlns   = (const float*)d_in[31];
    const float* rlnb   = (const float*)d_in[32];
    const float* outw   = (const float*)d_in[33];
    const float* outb   = (const float*)d_in[34];
    float* out = (float*)d_out;

    // workspace layout
    char* p = (char*)d_ws;
    auto alloc = [&](size_t bytes) -> void* {
        void* r = (void*)p;
        p += (bytes + 255) & ~(size_t)255;
        return r;
    };
    u16* wqT   = (u16*)alloc((size_t)DD * DD * 2);
    u16* wkT   = (u16*)alloc((size_t)DD * DD * 2);
    u16* wvT   = (u16*)alloc((size_t)DD * DD * 2);
    u16* woT   = (u16*)alloc((size_t)DD * DD * 2);
    u16* ffw1T = (u16*)alloc((size_t)DD * MM * 2);
    u16* ffw2T = (u16*)alloc((size_t)MM * DD * 2);
    u16* rw1T  = (u16*)alloc((size_t)RR * DD * MM * 2);
    u16* rw2T  = (u16*)alloc((size_t)RR * MM * DD * 2);
    u16* outwT = (u16*)alloc((size_t)DD * VV * 2);
    float* xF  = (float*)alloc((size_t)BB * SS * DD * 4);
    u16* xb    = (u16*)alloc((size_t)BB * SS * DD * 2);
    u16* l2rn  = (u16*)alloc((size_t)BB * SS * DD * 2);
    u16* r2ln  = (u16*)alloc((size_t)BB * SS * DD * 2);
    u16* sumn  = (u16*)alloc((size_t)BB * SS * DD * 2);
    u16* qb    = (u16*)alloc((size_t)BB * SS * DD * 2);
    u16* xatt  = (u16*)alloc((size_t)BB * SS * DD * 2);
    u16* aen   = (u16*)alloc((size_t)BB * TP * DD * 2);
    u16* kbuf  = (u16*)alloc((size_t)BB * TP * DD * 2);
    u16* vtb   = (u16*)alloc((size_t)BB * TP * DD * 2);
    u16* hb    = (u16*)alloc((size_t)BB * SS * MM * 2);
    float* t1  = (float*)alloc((size_t)BB * MM * 4);
    float* tt  = (float*)alloc((size_t)BB * 4 * DD * 4);
    float* film= (float*)alloc((size_t)RR * BB * 2 * DD * 4);

    dim3 blk256(256);
    // weight prep: transpose to N-major bf16 (done every call; deterministic)
    transpose_bf16_k<<<dim3(DD / 32, DD / 32), blk256, 0, stream>>>(Wq, wqT, DD, DD);
    transpose_bf16_k<<<dim3(DD / 32, DD / 32), blk256, 0, stream>>>(Wk, wkT, DD, DD);
    transpose_bf16_k<<<dim3(DD / 32, DD / 32), blk256, 0, stream>>>(Wv, wvT, DD, DD);
    transpose_bf16_k<<<dim3(DD / 32, DD / 32), blk256, 0, stream>>>(Wo, woT, DD, DD);
    transpose_bf16_k<<<dim3(MM / 32, DD / 32), blk256, 0, stream>>>(ffw1, ffw1T, DD, MM);
    transpose_bf16_k<<<dim3(DD / 32, MM / 32), blk256, 0, stream>>>(ffw2, ffw2T, MM, DD);
    for (int i = 0; i < RR; i++) {
        transpose_bf16_k<<<dim3(MM / 32, DD / 32), blk256, 0, stream>>>(
            rw1 + (size_t)i * DD * MM, rw1T + (size_t)i * DD * MM, DD, MM);
        transpose_bf16_k<<<dim3(DD / 32, MM / 32), blk256, 0, stream>>>(
            rw2 + (size_t)i * MM * DD, rw2T + (size_t)i * MM * DD, MM, DD);
    }
    transpose_bf16_k<<<dim3(VV / 32, DD / 32), blk256, 0, stream>>>(outw, outwT, DD, VV);

    // dual prenorm + residual input
    ln_dual_k<<<dim3(BB * SS), blk256, 0, stream>>>(
        l2r, r2l, lnl2rs, lnl2rb, lnr2ls, lnr2lb, xF, l2rn, r2ln, sumn);

    // padded concat for K/V inputs
    build_all_e_k<<<dim3((BB * TP * DD) / 256), blk256, 0, stream>>>(temb, l2rn, r2ln, aen);

    // Q/K/V projections
    const float qscale = 0.10206207261596575f;  // 1/sqrt(96)
    launch_gemm(1, BB * SS, DD, DD, stream, sumn, wqT, bq, nullptr, qb, nullptr, qscale, 0);
    launch_gemm(1, BB * TP, DD, DD, stream, aen, wkT, bk, nullptr, kbuf, nullptr, 1.0f, 0);
    launch_gemm(5, BB * TP, DD, DD, stream, aen, wvT, bv, nullptr, vtb, nullptr, 1.0f, TP);

    // flash attention
    flash_k<<<dim3(SS / 64, HH, BB), dim3(128), 0, stream>>>(qb, kbuf, vtb, xatt);

    // output projection + residual into x
    launch_gemm(4, BB * SS, DD, DD, stream, xatt, woT, bo, xF, nullptr, xF, 1.0f, 0);

    // FF block
    ln_bf16_k<<<dim3(BB * SS), blk256, 0, stream>>>(xF, ffls, fflb, xb);
    launch_gemm(2, BB * SS, MM, DD, stream, xb, ffw1T, ffb1, nullptr, hb, nullptr, 1.0f, 0);
    launch_gemm(4, BB * SS, DD, MM, stream, hb, ffw2T, ffb2, xF, xb, xF, 1.0f, 0);

    // temb MLP + FiLM params
    temb_mlp1_k<<<dim3((BB * MM) / 256), blk256, 0, stream>>>(temb, tmw1, tmb1, t1);
    temb_mlp2_k<<<dim3((BB * 4 * DD) / 256), blk256, 0, stream>>>(t1, tmw2, tmb2, tt);
    film_kern<<<dim3((RR * BB * 2 * DD) / 256), blk256, 0, stream>>>(tt, filmw, filmb, film);

    // residual readout blocks
    for (int i = 0; i < RR; i++) {
        launch_gemm(3, BB * SS, MM, DD, stream, xb, rw1T + (size_t)i * DD * MM,
                    rb1 + (size_t)i * MM, nullptr, hb, nullptr, 1.0f, 0);
        launch_gemm(4, BB * SS, DD, MM, stream, hb, rw2T + (size_t)i * MM * DD,
                    rb2 + (size_t)i * DD, xF, nullptr, xF, 1.0f, 0);
        ln_film_k<<<dim3(BB * SS), blk256, 0, stream>>>(
            xF, rlns + (size_t)i * DD, rlnb + (size_t)i * DD,
            film + (size_t)i * BB * 2 * DD, xb);
    }

    // vocab projection -> d_out (f32)
    launch_gemm(0, BB * SS, VV, DD, stream, xb, outwT, outb, out, nullptr, nullptr, 1.0f, 0);
}
